// RegionProposalNetwork_19456201851364
// MI455X (gfx1250) — matrix operand order repaired and checked by reading, not compiled, not run on hardware
//
#include <hip/hip_runtime.h>
#include <hip/hip_bf16.h>
#include <math.h>

// ---------------------------------------------------------------------------
// RPN for MI455X (gfx1250). Heavy compute = 3x3 conv as implicit GEMM via
// v_wmma_f32_16x16x32_bf16 (wave32), double-buffered LDS pipeline, and (when
// the toolchain exposes it) GLOBAL_LOAD_ASYNC_TO_LDS_B128 + s_wait_asynccnt.
// ---------------------------------------------------------------------------

typedef __attribute__((ext_vector_type(16))) __bf16 v16bf;
typedef __attribute__((ext_vector_type(8)))  float  v8f;

#define HH      50
#define WW      63
#define HW      3150          // 50*63
#define NPAD    3200          // N padded to 64-tile multiple
#define CIN     512
#define MOUT    512
#define KTOT    4608          // 512*9, = 144 * 32
#define NA      9
#define NANCH   28350         // HW*NA
#define NSORT   32768
#define NPRE    12000
#define NPOST   2000
#define NWORDS  375           // 12000/32
// d_out element offsets (float/int32 slots)
#define O_LOCS    0
#define O_SCORES  113400
#define O_ROIS    170100
#define O_RIDX    178100
#define O_CNT     180100
#define O_ANCH    180101

#if defined(__gfx1250__)
#if __has_builtin(__builtin_amdgcn_global_load_async_to_lds_b128) && \
    __has_builtin(__builtin_amdgcn_s_wait_asynccnt)
#define HAVE_ASYNC_LDS 1
#endif
#endif

__device__ __forceinline__ unsigned short f2bf(float f) {
    unsigned u = __float_as_uint(f);
    u += 0x7FFFu + ((u >> 16) & 1u);      // round-to-nearest-even
    return (unsigned short)(u >> 16);
}

// ---- 1. weights fp32 -> bf16 (layout (M,C,3,3) == (M, K) row-major) -------
__global__ void k_w2bf(const float* __restrict__ w, unsigned short* __restrict__ wb, int n) {
    int i = blockIdx.x * 256 + threadIdx.x;
    if (i < n) wb[i] = f2bf(w[i]);
}

// ---- 2. im2col: Xc[n][k] bf16, n in [0,3200) (rows >= 3150 zeroed) --------
__global__ void k_im2col(const float* __restrict__ x, unsigned short* __restrict__ xc) {
    int id = blockIdx.x * 256 + threadIdx.x;        // grid covers 3200*4608 exactly
    int n = id / KTOT, k = id - n * KTOT;
    float v = 0.f;
    if (n < HW) {
        int c = k / 9, r = k - c * 9;
        int ky = r / 3, kx = r - ky * 3;
        int y = n / WW, xx = n - y * WW;
        int iy = y + ky - 1, ix = xx + kx - 1;
        if (iy >= 0 && iy < HH && ix >= 0 && ix < WW)
            v = x[(c * HH + iy) * WW + ix];
    }
    xc[id] = f2bf(v);
}

// ---- 3. GEMM 512x3200x4608 bf16->f32 with bias+ReLU, WMMA ------------------
// block tile 128(M) x 64(N), 256 threads = 8 waves (4x2), wave tile 32x32,
// double-buffered LDS so global fills overlap WMMA compute.
#define LDSA_STRIDE 40   // halfwords per LDS row (32 data + 8 pad), 80B = 5 uint4
#define NKC (KTOT / 32)
__global__ __launch_bounds__(256)
void k_gemm(const unsigned short* __restrict__ Wb,
            const unsigned short* __restrict__ Xc,
            const float* __restrict__ bias,
            float* __restrict__ H) {
    __shared__ __align__(16) unsigned short As[2][128 * LDSA_STRIDE];
    __shared__ __align__(16) unsigned short Bs[2][64  * LDSA_STRIDE];
    const int tid  = threadIdx.x;
    const int bM   = blockIdx.y * 128;
    const int bN   = blockIdx.x * 64;
    const int wid  = tid >> 5, lane = tid & 31;
    const int wm   = wid & 3,  wn   = wid >> 2;
    const int l16  = lane & 15, lh  = lane >> 4;

    v8f acc[2][2];
    v8f zero = {};
    acc[0][0] = zero; acc[0][1] = zero; acc[1][0] = zero; acc[1][1] = zero;

    // global tile sources (uint4 = 8 bf16); row stride 4608/8 = 576 uint4
    const uint4* gA = (const uint4*)Wb;
    const uint4* gB = (const uint4*)Xc;
    // per-thread fill assignment: A = 512 16B chunks (2/thread), B = 256 (1/thread)
    const int rA0 = tid >> 2,        sgA0 = tid & 3;
    const int rA1 = (tid + 256) >> 2, sgA1 = tid & 3;   // (tid+256)&3 == tid&3
    const int nB  = tid >> 2,        sgB  = tid & 3;
    const size_t idxA0 = (size_t)(bM + rA0) * 576 + sgA0;
    const size_t idxA1 = (size_t)(bM + rA1) * 576 + sgA1;
    const size_t idxB  = (size_t)(bN + nB ) * 576 + sgB;
    const int ldsA0 = rA0 * 5 + sgA0;   // uint4 index inside one A buffer
    const int ldsA1 = rA1 * 5 + sgA1;
    const int ldsB  = nB  * 5 + sgB;

    // per-thread fragment offsets (halfword units inside one buffer)
    const int offA0 = (wm * 32 + l16) * LDSA_STRIDE + lh * 8;
    const int offA1 = offA0 + 16 * LDSA_STRIDE;
    const int offB0 = (wn * 32 + l16) * LDSA_STRIDE + lh * 8;
    const int offB1 = offB0 + 16 * LDSA_STRIDE;

    auto compute = [&](int cur) {
        union Fr { uint4 q[2]; v16bf v; } a0, a1, b0, b1;
        const unsigned short* Ab = As[cur];
        const unsigned short* Bb = Bs[cur];
        a0.q[0] = *(const uint4*)&Ab[offA0];
        a0.q[1] = *(const uint4*)&Ab[offA0 + 16];
        a1.q[0] = *(const uint4*)&Ab[offA1];
        a1.q[1] = *(const uint4*)&Ab[offA1 + 16];
        b0.q[0] = *(const uint4*)&Bb[offB0];
        b0.q[1] = *(const uint4*)&Bb[offB0 + 16];
        b1.q[0] = *(const uint4*)&Bb[offB1];
        b1.q[1] = *(const uint4*)&Bb[offB1 + 16];
        acc[0][0] = __builtin_amdgcn_wmma_f32_16x16x32_bf16(false, a0.v, false, b0.v, (short)0, acc[0][0], false, false);
        acc[0][1] = __builtin_amdgcn_wmma_f32_16x16x32_bf16(false, a0.v, false, b1.v, (short)0, acc[0][1], false, false);
        acc[1][0] = __builtin_amdgcn_wmma_f32_16x16x32_bf16(false, a1.v, false, b0.v, (short)0, acc[1][0], false, false);
        acc[1][1] = __builtin_amdgcn_wmma_f32_16x16x32_bf16(false, a1.v, false, b1.v, (short)0, acc[1][1], false, false);
    };

#ifdef HAVE_ASYNC_LDS
    // CDNA5 async copy: GLOBAL_LOAD_ASYNC_TO_LDS_B128, tracked by ASYNCcnt.
    // Builtin signature (from clang diagnostics): (v4i AS1* src, v4i AS3* dst,
    // imm offset, imm cpol).
    typedef int async_v4i __attribute__((vector_size(16)));
    typedef __attribute__((address_space(1))) async_v4i* gvp;
    typedef __attribute__((address_space(3))) async_v4i* lvp;
    auto issue = [&](int buf, int k8) {
        uint4* dA = (uint4*)As[buf];
        uint4* dB = (uint4*)Bs[buf];
        __builtin_amdgcn_global_load_async_to_lds_b128((gvp)(gA + idxA0 + k8), (lvp)(dA + ldsA0), 0, 0);
        __builtin_amdgcn_global_load_async_to_lds_b128((gvp)(gA + idxA1 + k8), (lvp)(dA + ldsA1), 0, 0);
        __builtin_amdgcn_global_load_async_to_lds_b128((gvp)(gB + idxB  + k8), (lvp)(dB + ldsB ), 0, 0);
    };
    issue(0, 0);
    __builtin_amdgcn_s_wait_asynccnt(0);
    __syncthreads();
    for (int kc = 0; kc < NKC; ++kc) {
        const int cur = kc & 1;
        if (kc + 1 < NKC) issue(cur ^ 1, (kc + 1) * 4);   // fill next buf async
        compute(cur);                                     // overlap with fills
        __builtin_amdgcn_s_wait_asynccnt(0);
        __syncthreads();
    }
#else
    // fallback: register-staged double buffer (still overlaps load & compute)
    uint4 pA0 = gA[idxA0], pA1 = gA[idxA1], pB = gB[idxB];
    { uint4* dA = (uint4*)As[0]; uint4* dB = (uint4*)Bs[0];
      dA[ldsA0] = pA0; dA[ldsA1] = pA1; dB[ldsB] = pB; }
    __syncthreads();
    for (int kc = 0; kc < NKC; ++kc) {
        const int cur = kc & 1;
        if (kc + 1 < NKC) {
            const int k8 = (kc + 1) * 4;
            pA0 = gA[idxA0 + k8]; pA1 = gA[idxA1 + k8]; pB = gB[idxB + k8];
        }
        compute(cur);
        if (kc + 1 < NKC) {
            uint4* dA = (uint4*)As[cur ^ 1]; uint4* dB = (uint4*)Bs[cur ^ 1];
            dA[ldsA0] = pA0; dA[ldsA1] = pA1; dB[ldsB] = pB;
        }
        __syncthreads();
    }
#endif

    // epilogue: C/D layout -> lane n = n0+l16, row m = m0 + r + 8*lh
    for (int ti = 0; ti < 2; ++ti)
        for (int tj = 0; tj < 2; ++tj) {
            int m0 = bM + wm * 32 + ti * 16;
            int nc = bN + wn * 32 + tj * 16 + l16;
            #pragma unroll
            for (int r = 0; r < 8; ++r) {
                int m = m0 + r + 8 * lh;
                float v = acc[ti][tj][r] + bias[m];
                H[(size_t)m * NPAD + nc] = v > 0.f ? v : 0.f;
            }
        }
}

// ---- 4. 1x1 heads + softmax pos score -------------------------------------
__global__ __launch_bounds__(64)
void k_heads(const float* __restrict__ H,
             const float* __restrict__ wsc, const float* __restrict__ bsc,
             const float* __restrict__ wlc, const float* __restrict__ blc,
             float* __restrict__ out_locs, float* __restrict__ out_scores,
             float* __restrict__ pos) {
    int n = blockIdx.x, t = threadIdx.x;
    __shared__ float hcol[512];
    __shared__ float outv[54];
    for (int c = t; c < 512; c += 64) hcol[c] = H[(size_t)c * NPAD + n];
    __syncthreads();
    if (t < 54) {
        const float* w = (t < 18) ? (wsc + t * 512) : (wlc + (t - 18) * 512);
        float s = 0.f;
        for (int c = 0; c < 512; ++c) s += w[c] * hcol[c];
        s += (t < 18) ? bsc[t] : blc[t - 18];
        outv[t] = s;
    }
    __syncthreads();
    if (t < 36) { int a = t >> 2; out_locs[((size_t)n * NA + a) * 4 + (t & 3)] = outv[18 + t]; }
    if (t < 18) { int a = t >> 1; out_scores[((size_t)n * NA + a) * 2 + (t & 1)] = outv[t]; }
    if (t < 9) {
        float s0 = outv[2 * t], s1 = outv[2 * t + 1];
        float m = fmaxf(s0, s1);
        float e0 = __expf(s0 - m), e1 = __expf(s1 - m);
        pos[(size_t)n * NA + t] = e1 / (e0 + e1);
    }
}

// ---- 5. anchors + loc2bbox + clip + valid mask + sort keys ----------------
__global__ void k_prop(const float* __restrict__ locs, const float* __restrict__ pos,
                       const int* __restrict__ imgh, const int* __restrict__ imgw,
                       float* __restrict__ anch_out, float* __restrict__ roi,
                       unsigned long long* __restrict__ keys) {
    int i = blockIdx.x * 256 + threadIdx.x;   // grid covers NSORT exactly
    if (i >= NANCH) { if (i < NSORT) keys[i] = ~0ULL; return; }
    int n = i / NA, a = i - n * NA;
    int ri = a / 3, si = a - ri * 3;
    float r = 0.5f * (float)(1 << ri);        // 0.5, 1, 2
    float s = (float)(8 << si);               // 8, 16, 32
    float hb = 16.f * s * sqrtf(r);
    float wb = 16.f * s * sqrtf(1.f / r);
    int y = n / WW, x = n - y * WW;
    float sy = y * 16.f, sx = x * 16.f;
    float a0 = 8.f - hb * 0.5f + sy, a1 = 8.f - wb * 0.5f + sx;
    float a2 = 8.f + hb * 0.5f + sy, a3 = 8.f + wb * 0.5f + sx;
    anch_out[i * 4 + 0] = a0; anch_out[i * 4 + 1] = a1;
    anch_out[i * 4 + 2] = a2; anch_out[i * 4 + 3] = a3;

    float ha = a2 - a0, wa = a3 - a1;
    float cy = a0 + 0.5f * ha, cx = a1 + 0.5f * wa;
    float dy = locs[i * 4 + 0], dx = locs[i * 4 + 1];
    float dh = locs[i * 4 + 2], dw = locs[i * 4 + 3];
    float ncy = dy * ha + cy, ncx = dx * wa + cx;
    float nh = expf(dh) * ha, nw = expf(dw) * wa;
    float fh = (float)imgh[0], fw = (float)imgw[0];
    float r0 = fminf(fmaxf(ncy - 0.5f * nh, 0.f), fh);
    float r1 = fminf(fmaxf(ncx - 0.5f * nw, 0.f), fw);
    float r2 = fminf(fmaxf(ncy + 0.5f * nh, 0.f), fh);
    float r3 = fminf(fmaxf(ncx + 0.5f * nw, 0.f), fw);
    roi[i * 4 + 0] = r0; roi[i * 4 + 1] = r1; roi[i * 4 + 2] = r2; roi[i * 4 + 3] = r3;

    bool valid = (r2 - r0 >= 16.f) && (r3 - r1 >= 16.f);
    float sc = valid ? pos[i] : -__builtin_inff();
    unsigned u = __float_as_uint(sc);
    unsigned mapped = (u & 0x80000000u) ? ~u : (u | 0x80000000u);  // order-preserving
    unsigned hi = ~mapped;                                          // ascending key = score desc
    keys[i] = ((unsigned long long)hi << 32) | (unsigned)i;         // tie-break: index asc
}

// ---- 6. bitonic sort pass (ascending) -------------------------------------
__global__ void k_bitonic(unsigned long long* __restrict__ keys, int j, int k) {
    int i = blockIdx.x * 256 + threadIdx.x;
    int ixj = i ^ j;
    if (ixj > i) {
        unsigned long long a = keys[i], b = keys[ixj];
        bool up = ((i & k) == 0);
        if ((a > b) == up) { keys[i] = b; keys[ixj] = a; }
    }
}

// ---- 7. gather top-12000 boxes (zero non-finite) --------------------------
__global__ void k_gather(const unsigned long long* __restrict__ keys,
                         const float* __restrict__ roi, float* __restrict__ rs) {
    int t = blockIdx.x * 256 + threadIdx.x;
    if (t >= NPRE) return;
    unsigned long long kk = keys[t];
    unsigned hi = (unsigned)(kk >> 32);
    unsigned idx = (unsigned)kk;
    bool ok = (hi != 0xFF800000u);   // masked (-inf) entries -> zero box
    #pragma unroll
    for (int c = 0; c < 4; ++c) rs[(size_t)t * 4 + c] = ok ? roi[(size_t)idx * 4 + c] : 0.f;
}

// ---- 8. IoU suppression bitmasks ------------------------------------------
__global__ void k_mask(const float* __restrict__ rs, unsigned* __restrict__ mask) {
    int id = blockIdx.x * 256 + threadIdx.x;
    if (id >= NPRE * NWORDS) return;
    int i = id / NWORDS, w = id - i * NWORDS;
    float y1 = rs[i * 4 + 0], x1 = rs[i * 4 + 1], y2 = rs[i * 4 + 2], x2 = rs[i * 4 + 3];
    float ai = fmaxf(y2 - y1, 0.f) * fmaxf(x2 - x1, 0.f);
    unsigned bits = 0;
    int jb = w * 32;
    for (int b = 0; b < 32; ++b) {
        int j = jb + b;
        float jy1 = rs[j * 4 + 0], jx1 = rs[j * 4 + 1];
        float jy2 = rs[j * 4 + 2], jx2 = rs[j * 4 + 3];
        float aj = fmaxf(jy2 - jy1, 0.f) * fmaxf(jx2 - jx1, 0.f);
        float iy = fmaxf(fminf(y2, jy2) - fmaxf(y1, jy1), 0.f);
        float ix = fmaxf(fminf(x2, jx2) - fmaxf(x1, jx1), 0.f);
        float inter = iy * ix;
        float iou = inter / fmaxf(ai + aj - inter, 1e-10f);
        if (iou > 0.7f) bits |= (1u << b);
    }
    mask[id] = bits;
}

// ---- 9. sequential greedy NMS (single block) ------------------------------
__global__ __launch_bounds__(512)
void k_nms(const unsigned* __restrict__ mask, int* __restrict__ keep_out, int* __restrict__ cnt_out) {
    __shared__ unsigned alive[NWORDS];
    __shared__ int keepS[NPOST];
    __shared__ int s_flag, s_cnt;
    int t = threadIdx.x;
    for (int w = t; w < NWORDS; w += 512) alive[w] = 0xFFFFFFFFu;
    for (int k = t; k < NPOST; k += 512) keepS[k] = -1;
    if (t == 0) s_cnt = 0;
    __syncthreads();
    for (int i = 0; i < NPRE; ++i) {
        if (t == 0) {
            int al = (alive[i >> 5] >> (i & 31)) & 1;
            int kf = (al && s_cnt < NPOST) ? 1 : 0;
            if (kf) { keepS[s_cnt] = i; s_cnt++; }
            s_flag = kf;
        }
        __syncthreads();
        if (s_flag) {
            const unsigned* row = mask + (size_t)i * NWORDS;
            for (int w = t; w < NWORDS; w += 512) alive[w] &= ~row[w];
        }
        __syncthreads();
    }
    for (int k = t; k < NPOST; k += 512) keep_out[k] = keepS[k];
    if (t == 0) cnt_out[0] = s_cnt;
}

// ---- 10. finalize rois / roi_indices / counts -----------------------------
__global__ void k_final(const int* __restrict__ keep, const int* __restrict__ cnt,
                        const float* __restrict__ rs, float* __restrict__ dout) {
    int k = blockIdx.x * 256 + threadIdx.x;
    if (k < NPOST) {
        int ki = keep[k];
        #pragma unroll
        for (int c = 0; c < 4; ++c)
            dout[O_ROIS + (size_t)k * 4 + c] = (ki >= 0) ? rs[(size_t)ki * 4 + c] : 0.f;
        ((int*)dout)[O_RIDX + k] = 0;            // n==1 -> all zeros (int32)
    }
    if (k == 0) ((int*)dout)[O_CNT] = cnt[0];    // int32
}

// ---------------------------------------------------------------------------
extern "C" void kernel_launch(void* const* d_in, const int* in_sizes, int n_in,
                              void* d_out, int out_size, void* d_ws, size_t ws_size,
                              hipStream_t stream) {
    const float* x      = (const float*)d_in[0];
    const float* w_conv = (const float*)d_in[1];
    const float* b_conv = (const float*)d_in[2];
    const float* w_sc   = (const float*)d_in[3];
    const float* b_sc   = (const float*)d_in[4];
    const float* w_lc   = (const float*)d_in[5];
    const float* b_lc   = (const float*)d_in[6];
    const int*   imgh   = (const int*)d_in[7];
    const int*   imgw   = (const int*)d_in[8];
    float* out = (float*)d_out;

    // workspace layout (bytes). post-GEMM scratch aliases the dead im2col buf.
    char* ws = (char*)d_ws;
    unsigned short* Wb = (unsigned short*)(ws);                    //  4,718,592 B
    unsigned short* Xc = (unsigned short*)(ws + 4718592);          // 29,491,200 B
    float* Hb  = (float*)(ws + 4718592 + 29491200);                //  6,553,600 B
    float* pos = (float*)(ws + 40763392);                          //    113,664 B
    float* roi = (float*)(ws + 40877056);                          //    453,632 B
    // aliased into Xc region (only used after GEMM):
    unsigned* maskp = (unsigned*)(ws + 4718592);                   // 18,000,000 B
    unsigned long long* keys = (unsigned long long*)(ws + 4718592 + 18000000);
    float* rois_s = (float*)(ws + 4718592 + 18262144);
    int* keepw = (int*)(ws + 4718592 + 18454144);
    int* cntw  = (int*)(ws + 4718592 + 18462144);

    // 1. weights -> bf16
    k_w2bf<<<(MOUT * KTOT + 255) / 256, 256, 0, stream>>>(w_conv, Wb, MOUT * KTOT);
    // 2. im2col -> bf16 (N padded to 3200)
    k_im2col<<<(NPAD * KTOT) / 256, 256, 0, stream>>>(x, Xc);
    // 3. WMMA GEMM + bias + ReLU
    k_gemm<<<dim3(NPAD / 64, MOUT / 128), 256, 0, stream>>>(Wb, Xc, b_conv, Hb);
    // 4. heads + softmax; writes rpn_locs / rpn_scores into d_out directly
    k_heads<<<HW, 64, 0, stream>>>(Hb, w_sc, b_sc, w_lc, b_lc,
                                   out + O_LOCS, out + O_SCORES, pos);
    // 5. anchors + loc2bbox + keys (writes anchor output)
    k_prop<<<NSORT / 256, 256, 0, stream>>>(out + O_LOCS, pos, imgh, imgw,
                                            out + O_ANCH, roi, keys);
    // 6. exact stable top-k via full bitonic sort of 32768 keys
    for (int k = 2; k <= NSORT; k <<= 1)
        for (int j = k >> 1; j > 0; j >>= 1)
            k_bitonic<<<NSORT / 256, 256, 0, stream>>>(keys, j, k);
    // 7. gather top-12000 boxes
    k_gather<<<(NPRE + 255) / 256, 256, 0, stream>>>(keys, roi, rois_s);
    // 8. suppression bitmasks
    k_mask<<<(NPRE * NWORDS + 255) / 256, 256, 0, stream>>>(rois_s, maskp);
    // 9. sequential greedy NMS
    k_nms<<<1, 512, 0, stream>>>(maskp, keepw, cntw);
    // 10. finalize rois / roi_indices / counts
    k_final<<<(NPOST + 255) / 256, 256, 0, stream>>>(keepw, cntw, rois_s, out);
}
